// ImageEncoderViT_21895743275471
// MI455X (gfx1250) — matrix operand order repaired and checked by reading, not compiled, not run on hardware
//
#include <hip/hip_runtime.h>
#include <hip/hip_bf16.h>
#include <cstdint>
#include <cstddef>

// ---------------------------------------------------------------------------
// ViT window-attention block (SAM-style) for gfx1250 / MI455X.
// GEMMs: 128x128 LDS-tiled, double-buffered global_load_async_to_lds_b128
// staging (ASYNCcnt), compute via v_wmma_f32_16x16x32_f16 (f32 accumulate).
// ---------------------------------------------------------------------------

typedef __attribute__((ext_vector_type(16))) _Float16 v16h;
typedef __attribute__((ext_vector_type(8)))  float    v8f;

#define DEV __device__ __forceinline__

static constexpr int   DIMC   = 768;
static constexpr int   NHEAD  = 12;
static constexpr int   HD     = 64;
static constexpr int   WS     = 14;
static constexpr int   NTOK   = WS * WS;          // 196 tokens / window
static constexpr int   NWIN   = 8 * 5 * 5;        // 200 windows
static constexpr int   MW     = NWIN * NTOK;      // 39200 windowed tokens
static constexpr int   MWP    = 39296;            // padded to 307*128
static constexpr int   MT     = 8 * 64 * 64;      // 32768 real tokens (256*128)
static constexpr int   KPAD   = 224;              // attn K padded to 7*32
static constexpr float SCALE  = 0.125f;           // 64^-0.5

DEV v8f wmma16(v16h a, v16h b, v8f c) {
  return __builtin_amdgcn_wmma_f32_16x16x32_f16(false, a, false, b,
                                                (short)0, c, false, false);
}

// A fragment (16x32, row-major source, ld in halves):
// lane L holds row M=L&15; halves 0..15 = K kb..kb+15,
// kb = k0 + (L>=16 ? 16 : 0).  Two 16-byte loads.
DEV v16h load_a_frag(const _Float16* A, int row, int ld, int k0, int lane) {
  int kb = k0 + ((lane & 16) ? 16 : 0);
  const _Float16* p = A + (size_t)row * ld + kb;
  union { v16h v; uint4 u[2]; } r;
  r.u[0] = *(const uint4*)(p);
  r.u[1] = *(const uint4*)(p + 8);
  return r.v;
}

// B fragment (32x16) from B^T stored row-major (N x K):
// lane L holds column N=L&15; 16 contiguous halves at K = k0 + (L>=16 ? 16 : 0).
DEV v16h load_b_frag(const _Float16* BT, int col, int ld, int k0, int lane) {
  int kb = k0 + ((lane & 16) ? 16 : 0);
  const _Float16* p = BT + (size_t)col * ld + kb;
  union { v16h v; uint4 u[2]; } r;
  r.u[0] = *(const uint4*)(p);
  r.u[1] = *(const uint4*)(p + 8);
  return r.v;
}

// gfx1250 async copy: global -> LDS, 16 bytes, tracked by ASYNCcnt.
DEV void async_copy_b128(uint32_t lds_off, const _Float16* gp) {
  asm volatile("global_load_async_to_lds_b128 %0, %1, off"
               :: "v"(lds_off), "v"((uint64_t)(uintptr_t)gp)
               : "memory");
}
DEV void wait_asynccnt0() {
  asm volatile("s_wait_asynccnt 0x0" ::: "memory");
}
DEV uint32_t lds_off32(const void* p) {   // low 32 bits of generic = LDS offset
  return (uint32_t)(uintptr_t)p;
}

DEV float gelu_exact(float x) {
  return 0.5f * x * (1.0f + erff(x * 0.70710678118654752f));
}

// --------------------------- weight convert --------------------------------
// src: f32 (K x N) row-major  ->  dst: f16 (N x K) row-major (B^T layout)
__global__ void cvt_w_kernel(const float* __restrict__ src, _Float16* __restrict__ dst,
                             int K, int N) {
  size_t i = (size_t)blockIdx.x * 256 + threadIdx.x;
  if (i >= (size_t)K * N) return;
  int n = (int)(i % N), k = (int)(i / N);
  dst[(size_t)n * K + k] = (_Float16)src[i];
}

__global__ void fill0_f16(_Float16* __restrict__ p, size_t n) {
  size_t i = (size_t)blockIdx.x * 256 + threadIdx.x;
  if (i < n) p[i] = (_Float16)0.f;
}

// --------------------------- LayerNorms ------------------------------------
__global__ void __launch_bounds__(256)
ln1_window_kernel(const float* __restrict__ x, const float* __restrict__ g,
                  const float* __restrict__ b, _Float16* __restrict__ xw) {
  int t    = blockIdx.x * 8 + (threadIdx.x >> 5);
  int lane = threadIdx.x & 31;
  int win = t / NTOK, nn = t % NTOK;
  int bb = win / 25, rem = win % 25, wh = rem / 5, ww = rem % 5;
  int i = nn / WS, j = nn % WS;
  int h = wh * WS + i, w = ww * WS + j;
  _Float16* dst = xw + (size_t)t * DIMC;
  if (h >= 64 || w >= 64) {          // zero-padded region (pad applied after LN)
    for (int c = lane; c < DIMC; c += 32) dst[c] = (_Float16)0.f;
    return;
  }
  const float* src = x + (((size_t)(bb * 64 + h)) * 64 + w) * DIMC;
  float s = 0.f, sq = 0.f;
  for (int c = lane; c < DIMC; c += 32) { float v = src[c]; s += v; sq += v * v; }
  for (int o = 16; o; o >>= 1) { s += __shfl_xor(s, o, 32); sq += __shfl_xor(sq, o, 32); }
  float mean = s * (1.f / DIMC);
  float var  = sq * (1.f / DIMC) - mean * mean;
  float rs   = rsqrtf(var + 1e-5f);
  for (int c = lane; c < DIMC; c += 32)
    dst[c] = (_Float16)((src[c] - mean) * rs * g[c] + b[c]);
}

__global__ void __launch_bounds__(256)
ln2_kernel(const float* __restrict__ x2, const float* __restrict__ g,
           const float* __restrict__ b, _Float16* __restrict__ xn2) {
  int t    = blockIdx.x * 8 + (threadIdx.x >> 5);
  int lane = threadIdx.x & 31;
  const float* src = x2 + (size_t)t * DIMC;
  _Float16*    dst = xn2 + (size_t)t * DIMC;
  float s = 0.f, sq = 0.f;
  for (int c = lane; c < DIMC; c += 32) { float v = src[c]; s += v; sq += v * v; }
  for (int o = 16; o; o >>= 1) { s += __shfl_xor(s, o, 32); sq += __shfl_xor(sq, o, 32); }
  float mean = s * (1.f / DIMC);
  float var  = sq * (1.f / DIMC) - mean * mean;
  float rs   = rsqrtf(var + 1e-5f);
  for (int c = lane; c < DIMC; c += 32)
    dst[c] = (_Float16)((src[c] - mean) * rs * g[c] + b[c]);
}

// --------------------------- LDS-tiled WMMA GEMM ---------------------------
// C(M x N) = A(M x K f16 rm) * B^T(N x K f16 rm) + bias, epilogue by EPI.
// Block: 256 threads = 8 waves (4 in M x 2 in N). Block tile 128x128.
// Double-buffered: async-copy tile k+1 into the spare LDS buffer while the
// 64 WMMAs of tile k execute. MG/NG enable M/N epilogue bounds guards.
struct EpiArgs {
  float*          out0;   // f32 output (x2 / d_out)
  const float*    resid;  // residual source
  _Float16*       out1;   // f16 output (gelu paths)
  _Float16*       q;      // qkv scatter targets
  _Float16*       k;
  _Float16*       vt;
};

template <int EPI, bool MG, bool NG>
__global__ void __launch_bounds__(256)
gemm_wmma(const _Float16* __restrict__ A, const _Float16* __restrict__ BT,
          const float* __restrict__ bias, int Mreal, int Nreal, int K, EpiArgs e) {
  __shared__ _Float16 sA[2][128 * 32];
  __shared__ _Float16 sB[2][128 * 32];
  const int tid  = threadIdx.x;
  const int lane = tid & 31;
  const int wave = tid >> 5;
  const int wm = (wave >> 1) * 32;        // 0,32,64,96
  const int wn = (wave & 1) * 64;         // 0,64
  const int m0b = blockIdx.y * 128;
  const int n0b = blockIdx.x * 128;

  // per-thread staging addresses (2 x 16B chunks of A, 2 of B per tile)
  const int ch0 = tid * 2, ch1 = tid * 2 + 1;  // 0..511
  const int r0 = ch0 >> 2, kc0 = (ch0 & 3) * 8;
  const int r1 = ch1 >> 2, kc1 = (ch1 & 3) * 8;
  const _Float16* gA0 = A  + (size_t)(m0b + r0) * K + kc0;
  const _Float16* gA1 = A  + (size_t)(m0b + r1) * K + kc1;
  const _Float16* gB0 = BT + (size_t)(n0b + r0) * K + kc0;
  const _Float16* gB1 = BT + (size_t)(n0b + r1) * K + kc1;
  const uint32_t lA0 = lds_off32(&sA[0][r0 * 32 + kc0]);
  const uint32_t lA1 = lds_off32(&sA[0][r1 * 32 + kc1]);
  const uint32_t lB0 = lds_off32(&sB[0][r0 * 32 + kc0]);
  const uint32_t lB1 = lds_off32(&sB[0][r1 * 32 + kc1]);
  const uint32_t BUFB = (uint32_t)(128 * 32 * 2);  // bytes between buffers

  auto stage = [&](int buf, int k0) {
    uint32_t bo = buf ? BUFB : 0u;
    async_copy_b128(lA0 + bo, gA0 + k0);
    async_copy_b128(lA1 + bo, gA1 + k0);
    async_copy_b128(lB0 + bo, gB0 + k0);
    async_copy_b128(lB1 + bo, gB1 + k0);
  };

  v8f acc[2][4];
#pragma unroll
  for (int i = 0; i < 2; ++i)
#pragma unroll
    for (int j = 0; j < 4; ++j) acc[i][j] = {};

  const int nk = K >> 5;
  stage(0, 0);
  for (int kt = 0; kt < nk; ++kt) {
    const int cur = kt & 1;
    wait_asynccnt0();     // my copies for tile kt have landed in LDS
    __syncthreads();      // all waves: tile kt staged; tile kt-1 fully consumed
    if (kt + 1 < nk) stage(cur ^ 1, (kt + 1) << 5);  // overlap with compute

    const _Float16* pA = sA[cur];
    const _Float16* pB = sB[cur];
    v16h a0 = load_a_frag(pA, wm + (lane & 15),      32, 0, lane);
    v16h a1 = load_a_frag(pA, wm + 16 + (lane & 15), 32, 0, lane);
#pragma unroll
    for (int j = 0; j < 4; ++j) {
      v16h bf = load_b_frag(pB, wn + j * 16 + (lane & 15), 32, 0, lane);
      acc[0][j] = wmma16(a0, bf, acc[0][j]);
      acc[1][j] = wmma16(a1, bf, acc[1][j]);
    }
  }

  // ------------------------- epilogue -------------------------
#pragma unroll
  for (int j = 0; j < 4; ++j) {
    const int n = n0b + wn + j * 16 + (lane & 15);
    if constexpr (NG) { if (n >= Nreal) continue; }
    const float bn = bias[n];
#pragma unroll
    for (int i = 0; i < 2; ++i) {
#pragma unroll
      for (int r = 0; r < 8; ++r) {
        const int m = m0b + wm + i * 16 + r + ((lane & 16) ? 8 : 0);
        if constexpr (MG) { if (m >= Mreal) continue; }
        const float v = acc[i][j][r] + bn;
        if constexpr (EPI == 0) {
          // QKV scatter: n = three*768 + head*64 + hd ; m = win*196 + nn
          int three = n / 768, cc = n % 768;
          int head = cc >> 6, hd = cc & 63;
          int win = m / NTOK, nn = m % NTOK;
          int bh = win * NHEAD + head;
          if (three == 0)      e.q[((size_t)bh * NTOK + nn) * HD + hd] = (_Float16)v;
          else if (three == 1) e.k[((size_t)bh * NTOK + nn) * HD + hd] = (_Float16)v;
          else                 e.vt[((size_t)bh * HD + hd) * KPAD + nn] = (_Float16)v;
        } else if constexpr (EPI == 1) {
          // GELU -> f16 row-major (M x Nreal)
          e.out1[(size_t)m * Nreal + n] = (_Float16)gelu_exact(v);
        } else if constexpr (EPI == 2) {
          // proj: window-unpartition + residual -> x2 (drop padded tokens)
          int win = m / NTOK, nn = m % NTOK;
          int bb = win / 25, rem = win % 25, wh = rem / 5, ww = rem % 5;
          int ii = nn / WS, jj = nn % WS;
          int hh = wh * WS + ii, wp = ww * WS + jj;
          if (hh < 64 && wp < 64) {
            size_t idx = (((size_t)(bb * 64 + hh)) * 64 + wp) * DIMC + n;
            e.out0[idx] = e.resid[idx] + v;
          }
        } else if constexpr (EPI == 3) {
          size_t idx = (size_t)m * DIMC + n;          // out = x2 + mlp
          e.out0[idx] = e.resid[idx] + v;
        } else {
          size_t idx = (size_t)m * DIMC + n;          // out += 0.5 * adapter
          e.out0[idx] += 0.5f * v;
        }
      }
    }
  }
}

// --------------------------- attention -------------------------------------
// One block (128 threads = 4 waves) per (window*head, 16-row q tile).
__global__ void __launch_bounds__(128)
attn_kernel(const _Float16* __restrict__ qb, const _Float16* __restrict__ kb,
            const _Float16* __restrict__ vt, const float* __restrict__ relh_tab,
            const float* __restrict__ relw_tab, _Float16* __restrict__ ob) {
  __shared__ float    s_scores[16 * 208];
  __shared__ _Float16 s_probs[16 * KPAD];
  __shared__ float    s_relh[16 * 16];
  __shared__ float    s_relw[16 * 16];

  const int bh   = blockIdx.y;            // 0..2399 (win*12 + head)
  const int m0   = blockIdx.x * 16;       // q-row tile base (0..192)
  const int lane = threadIdx.x & 31;
  const int wave = threadIdx.x >> 5;      // 0..3
  const int win  = bh / NHEAD, head = bh % NHEAD;

  const _Float16* q = qb + (size_t)bh * NTOK * HD;
  const _Float16* k = kb + (size_t)bh * NTOK * HD;
  const _Float16* v = vt + (size_t)bh * HD * KPAD;

  // ---- phase 0: decomposed rel-pos bias (uses UNscaled q) ----
  for (int idx = threadIdx.x; idx < 512; idx += 128) {
    int rr = idx >> 5, rem = idx & 31;
    int which = rem >> 4, kpos = rem & 15;
    float acc = 0.f;
    int nq = m0 + rr;
    if (kpos < WS && nq < NTOK) {
      int qh = nq / WS, qw = nq % WS;
      int pos = which ? qw : qh;
      const float* R = (which ? relw_tab : relh_tab) + (size_t)(pos - kpos + WS - 1) * HD;
      const _Float16* qp = q + (size_t)nq * HD;
#pragma unroll 8
      for (int c = 0; c < HD; ++c) acc += (float)qp[c] * R[c];
    }
    (which ? s_relw : s_relh)[rr * 16 + kpos] = acc;
  }
  __syncthreads();

  // ---- phase 1: scores = SCALE * q@k^T + rel_h + rel_w ----
  for (int tj = wave; tj < 13; tj += 4) {
    v8f acc = {};
    for (int k0 = 0; k0 < HD; k0 += 32) {
      v16h af = load_a_frag(q, m0 + (lane & 15), HD, k0, lane);
      v16h bf = load_b_frag(k, tj * 16 + (lane & 15), HD, k0, lane);
      acc = wmma16(af, bf, acc);
    }
#pragma unroll
    for (int r = 0; r < 8; ++r) {
      int rl = r + ((lane & 16) ? 8 : 0);
      int n  = tj * 16 + (lane & 15);
      s_scores[rl * 208 + n] =
          acc[r] * SCALE + s_relh[rl * 16 + n / WS] + s_relw[rl * 16 + n % WS];
    }
  }
  __syncthreads();

  // ---- phase 2: softmax-plus-one per row -> f16 probs (K padded w/ zeros) ----
  if (threadIdx.x < 16) {
    int t = threadIdx.x;
    if (m0 + t >= NTOK) {
      for (int jj = 0; jj < KPAD; ++jj) s_probs[t * KPAD + jj] = (_Float16)0.f;
    } else {
      float mx = -1e30f;
      for (int jj = 0; jj < NTOK; ++jj) mx = fmaxf(mx, s_scores[t * 208 + jj]);
      float sum = 0.f;
      for (int jj = 0; jj < NTOK; ++jj) sum += expf(s_scores[t * 208 + jj] - mx);
      float inv = 1.f / (expf(-mx) + sum);
      for (int jj = 0; jj < NTOK; ++jj)
        s_probs[t * KPAD + jj] = (_Float16)(expf(s_scores[t * 208 + jj] - mx) * inv);
      for (int jj = NTOK; jj < KPAD; ++jj) s_probs[t * KPAD + jj] = (_Float16)0.f;
    }
  }
  __syncthreads();

  // ---- phase 3: o = probs @ v ; wave -> 16 output channels ----
  {
    const int n0 = wave * 16;
    v8f acc = {};
    for (int k0 = 0; k0 < KPAD; k0 += 32) {
      v16h af = load_a_frag(s_probs, (lane & 15), KPAD, k0, lane);
      v16h bf = load_b_frag(v, n0 + (lane & 15), KPAD, k0, lane);
      acc = wmma16(af, bf, acc);
    }
#pragma unroll
    for (int r = 0; r < 8; ++r) {
      int rl = r + ((lane & 16) ? 8 : 0);
      int m  = m0 + rl;
      if (m < NTOK)
        ob[((size_t)win * NTOK + m) * DIMC + head * HD + n0 + (lane & 15)] = (_Float16)acc[r];
    }
  }
}

// --------------------------- launch ----------------------------------------
extern "C" void kernel_launch(void* const* d_in, const int* in_sizes, int n_in,
                              void* d_out, int out_size, void* d_ws, size_t ws_size,
                              hipStream_t stream) {
  const float* x      = (const float*)d_in[0];
  const float* ln1_g  = (const float*)d_in[1];
  const float* ln1_b  = (const float*)d_in[2];
  const float* qkv_w  = (const float*)d_in[3];
  const float* qkv_b  = (const float*)d_in[4];
  const float* proj_w = (const float*)d_in[5];
  const float* proj_b = (const float*)d_in[6];
  const float* relh   = (const float*)d_in[7];
  const float* relw   = (const float*)d_in[8];
  const float* ln2_g  = (const float*)d_in[9];
  const float* ln2_b  = (const float*)d_in[10];
  const float* mlp_w1 = (const float*)d_in[11];
  const float* mlp_b1 = (const float*)d_in[12];
  const float* mlp_w2 = (const float*)d_in[13];
  const float* mlp_b2 = (const float*)d_in[14];
  const float* ad_w1  = (const float*)d_in[15];
  const float* ad_b1  = (const float*)d_in[16];
  const float* ad_w2  = (const float*)d_in[17];
  const float* ad_b2  = (const float*)d_in[18];
  float* out = (float*)d_out;

  // workspace carve-up (256B aligned)
  size_t off = 0;
  auto alloc = [&](size_t bytes) -> char* {
    char* p = (char*)d_ws + off;
    off += (bytes + 255) & ~(size_t)255;
    return p;
  };
  _Float16* wq16 = (_Float16*)alloc((size_t)2304 * 768 * 2);
  _Float16* wp16 = (_Float16*)alloc((size_t)768 * 768 * 2);
  _Float16* wm1  = (_Float16*)alloc((size_t)3072 * 768 * 2);
  _Float16* wm2  = (_Float16*)alloc((size_t)768 * 3072 * 2);
  _Float16* wa1  = (_Float16*)alloc((size_t)256 * 768 * 2);   // N padded 192->256
  _Float16* wa2  = (_Float16*)alloc((size_t)768 * 192 * 2);
  _Float16* xw   = (_Float16*)alloc((size_t)MWP * DIMC * 2);  // M padded rows
  _Float16* qb   = (_Float16*)alloc((size_t)2400 * NTOK * HD * 2 + 4096); // tail pad
  _Float16* kb   = (_Float16*)alloc((size_t)2400 * NTOK * HD * 2 + 4096);
  _Float16* vt   = (_Float16*)alloc((size_t)2400 * HD * KPAD * 2);
  _Float16* ob   = (_Float16*)alloc((size_t)MWP * DIMC * 2);  // M padded rows
  float*    x2   = (float*)   alloc((size_t)MT * DIMC * 4);
  _Float16* xn2  = (_Float16*)alloc((size_t)MT * DIMC * 2);
  _Float16* h1   = (_Float16*)alloc((size_t)MT * 3072 * 2);
  _Float16* a1   = (_Float16*)alloc((size_t)MT * 192 * 2);

  auto cvt = [&](const float* s, _Float16* d, int K, int N) {
    size_t n = (size_t)K * N;
    cvt_w_kernel<<<dim3((unsigned)((n + 255) / 256)), 256, 0, stream>>>(s, d, K, N);
  };
  auto zfill = [&](_Float16* p, size_t n) {
    fill0_f16<<<dim3((unsigned)((n + 255) / 256)), 256, 0, stream>>>(p, n);
  };

  zfill(wa1, (size_t)256 * 768);          // pad rows must be benign
  zfill(vt,  (size_t)2400 * HD * KPAD);   // K-pad columns must be exact zero

  cvt(qkv_w,  wq16, 768, 2304);
  cvt(proj_w, wp16, 768, 768);
  cvt(mlp_w1, wm1,  768, 3072);
  cvt(mlp_w2, wm2,  3072, 768);
  cvt(ad_w1,  wa1,  768, 192);
  cvt(ad_w2,  wa2,  192, 768);

  // LN1 + window partition
  ln1_window_kernel<<<dim3(MW / 8), 256, 0, stream>>>(x, ln1_g, ln1_b, xw);

  // QKV (M padded rows of xw are garbage; epilogue guards m < MW)
  {
    EpiArgs e = {}; e.q = qb; e.k = kb; e.vt = vt;
    gemm_wmma<0, true, false><<<dim3(2304 / 128, MWP / 128), 256, 0, stream>>>(
        xw, wq16, qkv_b, MW, 2304, 768, e);
  }

  // attention
  attn_kernel<<<dim3(13, 2400), 128, 0, stream>>>(qb, kb, vt, relh, relw, ob);

  // proj + residual + window unpartition -> x2
  {
    EpiArgs e = {}; e.out0 = x2; e.resid = x;
    gemm_wmma<2, true, false><<<dim3(768 / 128, MWP / 128), 256, 0, stream>>>(
        ob, wp16, proj_b, MW, 768, 768, e);
  }

  // LN2
  ln2_kernel<<<dim3(MT / 8), 256, 0, stream>>>(x2, ln2_g, ln2_b, xn2);

  // MLP fc1 (GELU -> f16)
  {
    EpiArgs e = {}; e.out1 = h1;
    gemm_wmma<1, false, false><<<dim3(3072 / 128, MT / 128), 256, 0, stream>>>(
        xn2, wm1, mlp_b1, MT, 3072, 768, e);
  }
  // MLP fc2: out = x2 + mlp
  {
    EpiArgs e = {}; e.out0 = out; e.resid = x2;
    gemm_wmma<3, false, false><<<dim3(768 / 128, MT / 128), 256, 0, stream>>>(
        h1, wm2, mlp_b2, MT, 768, 3072, e);
  }
  // adapter fc1 (GELU -> f16, N padded 192->256 in weights/grid only)
  {
    EpiArgs e = {}; e.out1 = a1;
    gemm_wmma<1, false, true><<<dim3(256 / 128, MT / 128), 256, 0, stream>>>(
        xn2, wa1, ad_b1, MT, 192, 768, e);
  }
  // adapter fc2: out += 0.5 * ad
  {
    EpiArgs e = {}; e.out0 = out;
    gemm_wmma<4, false, false><<<dim3(768 / 128, MT / 128), 256, 0, stream>>>(
        a1, wa2, ad_b2, MT, 768, 192, e);
  }
}
